// GCNII_5188320494459
// MI455X (gfx1250) — matrix operand order repaired and checked by reading, not compiled, not run on hardware
//
#include <hip/hip_runtime.h>

// GCNII forward on gfx1250 (MI455X).
// Pipeline: degree/norm -> proj_in (WMMA f16) -> 8x [self-loop init, edge
// scatter-add (global_atomic_add_f32), fused-mix WMMA GEMM + relu] -> proj_out.
// h/x0/agg (3x51.2MB) live in workspace and stay L2-resident (192MB L2).

#define FEAT   128
#define ODIM   64
#define NLAY   8
#define ALPHA  0.1f

typedef __attribute__((ext_vector_type(16))) _Float16 v16h;
typedef __attribute__((ext_vector_type(8)))  float    v8f;

// ---------------- degree / normalization ----------------

__global__ void k_deg_init(float* deg, int n) {
    int i = blockIdx.x * 256 + threadIdx.x;
    if (i < n) deg[i] = 1.0f;  // self loop
}

__global__ void k_deg_edges(const int* __restrict__ col, float* deg, int E) {
    int e = blockIdx.x * 256 + threadIdx.x;
    if (e < E) unsafeAtomicAdd(&deg[col[e]], 1.0f);
}

__global__ void k_dinv(float* deg, int n) {
    int i = blockIdx.x * 256 + threadIdx.x;
    if (i < n) deg[i] = rsqrtf(deg[i]);  // deg >= 1 always (self loop)
}

// ---------------- weight conversion f32 -> f16 ----------------

__global__ void k_cvt_f16(const float* __restrict__ src, _Float16* __restrict__ dst, int n) {
    int i = blockIdx.x * 256 + threadIdx.x;
    if (i < n) dst[i] = (_Float16)src[i];
}

// ---------------- aggregation ----------------

// agg[i][:] = dinv[i]^2 * h[i][:]   (self-loop contribution), float4 per thread
__global__ void k_self_init(const float* __restrict__ h, const float* __restrict__ dinv,
                            float* __restrict__ agg, int n) {
    int i = blockIdx.x * 256 + threadIdx.x;
    if (i >= n * (FEAT / 4)) return;
    int node = i >> 5;
    int c = (i & 31) * 4;
    float d2 = dinv[node] * dinv[node];
    const float4 hv = *(const float4*)&h[(size_t)node * FEAT + c];
    float4 o = make_float4(d2 * hv.x, d2 * hv.y, d2 * hv.z, d2 * hv.w);
    *(float4*)&agg[(size_t)node * FEAT + c] = o;
}

// one wave per edge: agg[col] += dinv[row]*dinv[col] * h[row]
__global__ void k_edge_agg(const int* __restrict__ row, const int* __restrict__ col,
                           const float* __restrict__ dinv,
                           const float* __restrict__ h, float* __restrict__ agg, int E) {
    int e = blockIdx.x * 8 + (threadIdx.x >> 5);
    if (e >= E) return;
    int lane = threadIdx.x & 31;
    int r = row[e], c = col[e];
    float w = dinv[r] * dinv[c];
    const float* hs = h + (size_t)r * FEAT;
    float*       ad = agg + (size_t)c * FEAT;
#pragma unroll
    for (int f = 0; f < 4; ++f) {
        int idx = lane + 32 * f;
        unsafeAtomicAdd(&ad[idx], w * hs[idx]);
    }
}

// ---------------- WMMA GEMM helpers ----------------
// A fragment (16x32 f16): lane L -> row m=L%16; K-base kb = (L<16?0:8)+32*kk;
//   halves 0..7 = K kb..kb+7, halves 8..15 = K kb+16..kb+23 (contiguous 16B chunks).
// B fragment (32x16 f16): lane L -> col n = n0 + L%16; halves 0..7 = K kb..kb+7, halves 8..15 = K kb+16..kb+23 (same kb as A).
// C (16x16 f32): lane L -> n = L%16; VGPR r -> m = 8*(L>>4) + r.

__device__ __forceinline__ v16h load_a_frag(const _Float16* __restrict__ rowp) {
    v16h a;
#pragma unroll
    for (int j = 0; j < 8; ++j) { a[j] = rowp[j]; a[j + 8] = rowp[16 + j]; }
    return a;
}

__device__ __forceinline__ v16h load_b_frag(const _Float16* __restrict__ bp, int ld) {
    v16h b;
#pragma unroll
    for (int j = 0; j < 8; ++j) { b[j] = bp[(size_t)j * ld]; b[j + 8] = bp[(size_t)(16 + j) * ld]; }
    return b;
}

// ---------------- input projection: h = relu(x @ W_in + b_in); x0 = h ----------------

__global__ void __launch_bounds__(256)
k_gemm_in(const float* __restrict__ x, const _Float16* __restrict__ W,
          const float* __restrict__ bias, float* __restrict__ h, float* __restrict__ x0) {
    __shared__ _Float16 As[16][FEAT + 8];
    const int tid = threadIdx.x;
    const int m0  = blockIdx.x * 16;
    for (int idx = tid; idx < 16 * FEAT; idx += 256) {
        int r = idx >> 7, c = idx & (FEAT - 1);
        As[r][c] = (_Float16)x[(size_t)(m0 + r) * FEAT + c];
    }
    __syncthreads();
    const int wave = tid >> 5;          // n-tile 0..7
    const int lane = tid & 31;
    const int n0   = wave * 16;
    const int am   = lane & 15;
    const int akb  = (lane < 16) ? 0 : 8;
    v8f acc = {};
#pragma unroll
    for (int kk = 0; kk < 4; ++kk) {
        v16h a = load_a_frag(&As[am][kk * 32 + akb]);
        v16h b = load_b_frag(&W[(size_t)(kk * 32 + akb) * FEAT + n0 + am], FEAT);
        acc = __builtin_amdgcn_wmma_f32_16x16x32_f16(false, a, false, b,
                                                     (short)0, acc, false, false);
    }
    const int cn  = lane & 15;
    const int cm0 = (lane >> 4) * 8;
#pragma unroll
    for (int r = 0; r < 8; ++r) {
        float v = acc[r] + bias[n0 + cn];
        v = v > 0.0f ? v : 0.0f;
        size_t g = (size_t)(m0 + cm0 + r) * FEAT + n0 + cn;
        h[g]  = v;
        x0[g] = v;
    }
}

// ---------------- layer GEMM: h = relu(((1-a)*agg + a*x0) @ W) ----------------

__global__ void __launch_bounds__(256)
k_gemm_layer(const float* __restrict__ agg, const float* __restrict__ x0,
             const _Float16* __restrict__ W, float* __restrict__ h) {
    __shared__ _Float16 As[16][FEAT + 8];
    const int tid = threadIdx.x;
    const int m0  = blockIdx.x * 16;
    for (int idx = tid; idx < 16 * FEAT; idx += 256) {
        int r = idx >> 7, c = idx & (FEAT - 1);
        size_t g = (size_t)(m0 + r) * FEAT + c;
        As[r][c] = (_Float16)((1.0f - ALPHA) * agg[g] + ALPHA * x0[g]);
    }
    __syncthreads();
    const int wave = tid >> 5;
    const int lane = tid & 31;
    const int n0   = wave * 16;
    const int am   = lane & 15;
    const int akb  = (lane < 16) ? 0 : 8;
    v8f acc = {};
#pragma unroll
    for (int kk = 0; kk < 4; ++kk) {
        v16h a = load_a_frag(&As[am][kk * 32 + akb]);
        v16h b = load_b_frag(&W[(size_t)(kk * 32 + akb) * FEAT + n0 + am], FEAT);
        acc = __builtin_amdgcn_wmma_f32_16x16x32_f16(false, a, false, b,
                                                     (short)0, acc, false, false);
    }
    const int cn  = lane & 15;
    const int cm0 = (lane >> 4) * 8;
#pragma unroll
    for (int r = 0; r < 8; ++r) {
        float v = acc[r];
        v = v > 0.0f ? v : 0.0f;
        h[(size_t)(m0 + cm0 + r) * FEAT + n0 + cn] = v;
    }
}

// ---------------- output projection: out = h @ W_out + b_out ----------------

__global__ void __launch_bounds__(128)
k_gemm_out(const float* __restrict__ h, const _Float16* __restrict__ W,
           const float* __restrict__ bias, float* __restrict__ out) {
    __shared__ _Float16 As[16][FEAT + 8];
    const int tid = threadIdx.x;
    const int m0  = blockIdx.x * 16;
    for (int idx = tid; idx < 16 * FEAT; idx += 128) {
        int r = idx >> 7, c = idx & (FEAT - 1);
        As[r][c] = (_Float16)h[(size_t)(m0 + r) * FEAT + c];
    }
    __syncthreads();
    const int wave = tid >> 5;          // n-tile 0..3
    const int lane = tid & 31;
    const int n0   = wave * 16;
    const int am   = lane & 15;
    const int akb  = (lane < 16) ? 0 : 8;
    v8f acc = {};
#pragma unroll
    for (int kk = 0; kk < 4; ++kk) {
        v16h a = load_a_frag(&As[am][kk * 32 + akb]);
        v16h b = load_b_frag(&W[(size_t)(kk * 32 + akb) * ODIM + n0 + am], ODIM);
        acc = __builtin_amdgcn_wmma_f32_16x16x32_f16(false, a, false, b,
                                                     (short)0, acc, false, false);
    }
    const int cn  = lane & 15;
    const int cm0 = (lane >> 4) * 8;
#pragma unroll
    for (int r = 0; r < 8; ++r)
        out[(size_t)(m0 + cm0 + r) * ODIM + n0 + cn] = acc[r] + bias[n0 + cn];
}

// ---------------- host launcher ----------------

extern "C" void kernel_launch(void* const* d_in, const int* in_sizes, int n_in,
                              void* d_out, int out_size, void* d_ws, size_t ws_size,
                              hipStream_t stream) {
    const float* x      = (const float*)d_in[0];
    const int*   ei     = (const int*)d_in[1];     // int32 indices, shape (2, E)
    const float* W_in   = (const float*)d_in[2];
    const float* b_in   = (const float*)d_in[3];
    const float* conv_w = (const float*)d_in[4];
    const float* W_out  = (const float*)d_in[5];
    const float* b_out  = (const float*)d_in[6];
    float*       out    = (float*)d_out;

    const int N = in_sizes[0] / FEAT;   // 100000 (divisible by 16)
    const int E = in_sizes[1] / 2;      // 1600000
    const int* row = ei;
    const int* col = ei + E;

    // workspace carve-up (256B aligned)
    char* ws = (char*)d_ws;
    size_t off = 0;
    auto carve = [&](size_t bytes) -> char* {
        char* p = ws + off;
        off = (off + bytes + 255) & ~(size_t)255;
        return p;
    };
    float*    dinv   = (float*)carve((size_t)N * 4);
    float*    h      = (float*)carve((size_t)N * FEAT * 4);
    float*    x0     = (float*)carve((size_t)N * FEAT * 4);
    float*    agg    = (float*)carve((size_t)N * FEAT * 4);
    _Float16* Wf_in  = (_Float16*)carve((size_t)FEAT * FEAT * 2);
    _Float16* Wf_cv  = (_Float16*)carve((size_t)NLAY * FEAT * FEAT * 2);
    _Float16* Wf_out = (_Float16*)carve((size_t)FEAT * ODIM * 2);

    const int nb  = (N + 255) / 256;
    const int eb  = (E + 255) / 256;
    const int mt  = N / 16;             // 6250 m-tiles

    // 1) normalization: dinv = rsqrt(1 + in-degree)
    k_deg_init <<<nb, 256, 0, stream>>>(dinv, N);
    k_deg_edges<<<eb, 256, 0, stream>>>(col, dinv, E);
    k_dinv     <<<nb, 256, 0, stream>>>(dinv, N);

    // 2) weights -> f16
    k_cvt_f16<<<(FEAT * FEAT + 255) / 256, 256, 0, stream>>>(W_in, Wf_in, FEAT * FEAT);
    k_cvt_f16<<<(NLAY * FEAT * FEAT + 255) / 256, 256, 0, stream>>>(conv_w, Wf_cv,
                                                                    NLAY * FEAT * FEAT);
    k_cvt_f16<<<(FEAT * ODIM + 255) / 256, 256, 0, stream>>>(W_out, Wf_out, FEAT * ODIM);

    // 3) input projection
    k_gemm_in<<<mt, 256, 0, stream>>>(x, Wf_in, b_in, h, x0);

    // 4) GCNII layers
    const int sb = ((size_t)N * (FEAT / 4) + 255) / 256;
    const int ab = (E + 7) / 8;         // one wave per edge, 8 waves/block
    for (int l = 0; l < NLAY; ++l) {
        k_self_init<<<sb, 256, 0, stream>>>(h, dinv, agg, N);
        k_edge_agg <<<ab, 256, 0, stream>>>(row, col, dinv, h, agg, E);
        k_gemm_layer<<<mt, 256, 0, stream>>>(agg, x0, Wf_cv + (size_t)l * FEAT * FEAT, h);
    }

    // 5) output projection
    k_gemm_out<<<mt, 128, 0, stream>>>(h, Wf_out, b_out, out);
}
